// Quant_Conv_21440476742422
// MI455X (gfx1250) — matrix operand order repaired and checked by reading, not compiled, not run on hardware
//
#include <hip/hip_runtime.h>

typedef __attribute__((ext_vector_type(16))) _Float16 v16h;
typedef __attribute__((ext_vector_type(8)))  float    v8f;
typedef __attribute__((ext_vector_type(2)))  float    v2f;

// Problem: x [B=32,C=256,H=32,W=32] f32, w [O=256,C=256,3,3] f32, bias [256] f32
// -> out [32,256,32,32] f32.  GEMM view: M=32768, N=256, K=2304.
// float_quantize(exp=5, man=10) == fp16 RNE, so WMMA f32_16x16x32_f16 is exact-shaped.
#define BATCH 32
#define CIN   256
#define COUT  256
#define HW    1024          // 32*32
#define PH    34            // padded H/W (halo of 1)
#define XPAD_ELEMS ((size_t)BATCH * PH * PH * CIN)   // fp16 zero-padded NHWC

// ---------------------------------------------------------------------------
// Prep 1: x NCHW f32 -> zero-padded NHWC f16, channels kept in natural order
// (no swizzle); a lane's WMMA A-fragment is one
// contiguous 32-byte run:
//   lanes 0-15 take K 0..15, lanes 16-31 take K 16..31 (same order as B).
// ---------------------------------------------------------------------------
__global__ __launch_bounds__(1024)
void x_to_nhwc_pad_f16(const float* __restrict__ x, _Float16* __restrict__ xh) {
    __shared__ _Float16 tile[32][33];
    const int b  = blockIdx.z;
    const int c0 = blockIdx.y * 32;
    const int p0 = blockIdx.x * 32;
    const int tx = threadIdx.x;
    const int ty = threadIdx.y;
    float v = x[((size_t)b * CIN + (c0 + ty)) * HW + (p0 + tx)];
    tile[ty][tx] = (_Float16)v;
    __syncthreads();
    const int p = p0 + ty;
    const int h = p >> 5, w = p & 31;
    // channel tx -> same position within its 32-block (no swizzle)
    const int sw = tx;
    xh[(((size_t)b * PH + (h + 1)) * PH + (w + 1)) * CIN + (c0 + sw)] = tile[tx][ty];
}

// ---------------------------------------------------------------------------
// Prep 2: weight [O][C][3][3] f32 -> wh [tap][O][C] f16 (tap = kh*3+kw).
// B fragments use the same contiguous K as A (lanes lo: c0+0..15, hi: c0+16..31),
// so no swizzle here.
// ---------------------------------------------------------------------------
__global__ __launch_bounds__(256)
void w_to_taps_f16(const float* __restrict__ w, _Float16* __restrict__ wh) {
    const int idx = blockIdx.x * 256 + threadIdx.x;
    const int c   = idx & 255;
    const int o   = (idx >> 8) & 255;
    const int tap = idx >> 16;
    wh[idx] = (_Float16)w[((size_t)(o * CIN + c)) * 9 + tap];
}

// ---------------------------------------------------------------------------
// Main: implicit-GEMM conv, v_wmma_f32_16x16x32_f16.
//   256 threads = 8 waves; block tile 128(M) x 256(N); wave tile 64x64.
//   8 v16h loads -> 16 WMMA per k-chunk; all hot-loop offsets immediates.
// ---------------------------------------------------------------------------
__global__ __launch_bounds__(256)
void qconv_wmma(const _Float16* __restrict__ xh,
                const _Float16* __restrict__ wh,
                const float*    __restrict__ bias,
                float*          __restrict__ out) {
    __shared__ float swap[8][32 * 65];     // per-wave 32(n) x 64(m) f32, stride 65

    const int lane  = threadIdx.x & 31;
    const int wave  = threadIdx.x >> 5;
    const int waveM = wave & 1;            // 0..1
    const int waveN = wave >> 1;           // 0..3
    const int lhalf = lane & 15;
    const bool hi   = lane >= 16;

    const int m0 = blockIdx.x * 128 + waveM * 64;  // 64-aligned -> 2 image rows, same batch
    const int n0 = waveN * 64;                     // block covers all N=256
    const int b   = m0 >> 10;
    const int oh0 = (m0 >> 5) & 31;                // even

    // A: 4 M-fragment base pointers at tap (0,0); plain NHWC layout makes each
    // lane's fragment one contiguous v16h (lo lanes at +0, hi lanes at +16).
    const _Float16* ap[4];
#pragma unroll
    for (int f = 0; f < 4; ++f) {
        const int ohf = oh0 + (f >> 1);
        const int owf = ((f & 1) << 4) + lhalf;
        ap[f] = xh + (((size_t)b * PH + ohf) * PH + owf) * CIN + (hi ? 16 : 0);
    }
    // B: lanes 0-15 = col n, K c0+0..15; lanes 16-31 = col n, K c0+16..31.
    const _Float16* bp[4];
#pragma unroll
    for (int f = 0; f < 4; ++f)
        bp[f] = wh + ((size_t)(n0 + f * 16 + lhalf)) * CIN + (hi ? 16 : 0);

    v8f acc[4][4];
#pragma unroll
    for (int i = 0; i < 4; ++i)
#pragma unroll
        for (int j = 0; j < 4; ++j) acc[i][j] = (v8f){};

    for (int kh = 0; kh < 3; ++kh) {
#pragma unroll
        for (int kw = 0; kw < 3; ++kw) {
            const int ao = kw * CIN;               // immediate
            const int bo = kw * COUT * CIN;        // immediate
#pragma unroll
            for (int c0 = 0; c0 < CIN; c0 += 32) {
                v16h af[4], bf[4];
#pragma unroll
                for (int f = 0; f < 4; ++f) {
                    af[f] = *(const v16h*)(ap[f] + ao + c0);
                    bf[f] = *(const v16h*)(bp[f] + bo + c0);
                }
#pragma unroll
                for (int i = 0; i < 4; ++i)
#pragma unroll
                    for (int j = 0; j < 4; ++j)
                        acc[i][j] = __builtin_amdgcn_wmma_f32_16x16x32_f16(
                            false, af[i], false, bf[j], (short)0, acc[i][j],
                            false, false);
            }
        }
#pragma unroll
        for (int f = 0; f < 4; ++f) ap[f] += (size_t)PH * CIN;       // next kh row
#pragma unroll
        for (int f = 0; f < 4; ++f) bp[f] += (size_t)3 * COUT * CIN; // next 3 taps
    }

    // ---- Epilogue: quantize fp16, + bias, LDS transpose, coalesced b64 stores.
    float bv[4];
#pragma unroll
    for (int f = 0; f < 4; ++f) bv[f] = bias[n0 + f * 16 + lhalf];

    float* my = &swap[wave][0];
    // For fixed n, the 64 m-values are 64 consecutive f32 in NCHW starting here:
    float* ob = out + ((size_t)b * COUT + n0) * HW + oh0 * 32;
    const int mb = hi ? 8 : 0;

#pragma unroll
    for (int p = 0; p < 2; ++p) {      // two 32-channel passes reusing LDS tile
#pragma unroll
        for (int g = 0; g < 2; ++g) {
            const int fn = 2 * p + g;
            const int nl = g * 16 + lhalf;
#pragma unroll
            for (int fm = 0; fm < 4; ++fm)
#pragma unroll
                for (int r = 0; r < 8; ++r)
                    my[nl * 65 + fm * 16 + mb + r] =
                        (float)(_Float16)acc[fm][fn][r] + bv[fn];
        }
        // Same-wave LDS is processed in-order: pass-1 writes cannot pass pass-0 reads.
#pragma unroll
        for (int nl = 0; nl < 32; ++nl) {
            const v2f v = *(const v2f*)&my[nl * 65 + 2 * lane];
            *(v2f*)(ob + (size_t)(p * 32 + nl) * HW + 2 * lane) = v;
        }
    }
}

// ---------------------------------------------------------------------------
// Launch
// ---------------------------------------------------------------------------
extern "C" void kernel_launch(void* const* d_in, const int* in_sizes, int n_in,
                              void* d_out, int out_size, void* d_ws, size_t ws_size,
                              hipStream_t stream) {
    const float* x    = (const float*)d_in[0];
    const float* w    = (const float*)d_in[1];
    const float* bias = (const float*)d_in[2];
    float* out        = (float*)d_out;

    _Float16* xh = (_Float16*)d_ws;            // ~19 MB zero-padded NHWC f16
    _Float16* wh = xh + XPAD_ELEMS;            // ~1.2 MB [tap][O][C] f16

    hipMemsetAsync(xh, 0, XPAD_ELEMS * sizeof(_Float16), stream);
    x_to_nhwc_pad_f16<<<dim3(32, 8, BATCH), dim3(32, 32), 0, stream>>>(x, xh);
    w_to_taps_f16<<<dim3(9 * 256), dim3(256), 0, stream>>>(w, wh);

    // Main WMMA conv-GEMM: grid = M/128, block tile 128x256 covers all N.
    qconv_wmma<<<dim3(32768 / 128), dim3(256), 0, stream>>>(xh, wh, bias, out);
}
